// Deep_n_Cross_68917045232361
// MI455X (gfx1250) — matrix operand order repaired and checked by reading, not compiled, not run on hardware
//
#include <hip/hip_runtime.h>
#include <hip/hip_bf16.h>

typedef __bf16 bf16_t;
typedef __attribute__((ext_vector_type(16))) __bf16 v16bf;
typedef __attribute__((ext_vector_type(8)))  __bf16 v8bf;
typedef __attribute__((ext_vector_type(8)))  float  v8f;
typedef __attribute__((ext_vector_type(4)))  int    v4i;

#define BATCH  4096
#define NFIELD 26
#define EDIM   64
#define DDIM   1664   // 26*64
#define H1D    1024
#define H2D    512
#define H3D    256

// ---- gfx1250 async global->LDS path (guarded; falls back to ld/st) --------
#if defined(__has_builtin)
#if __has_builtin(__builtin_amdgcn_global_load_async_to_lds_b128)
#define USE_ASYNC_COPY 1
#endif
#endif
#ifndef USE_ASYNC_COPY
#define USE_ASYNC_COPY 0
#endif

#if defined(__has_builtin)
#if __has_builtin(__builtin_amdgcn_s_wait_asynccnt)
#define WAIT_ASYNCCNT(n) __builtin_amdgcn_s_wait_asynccnt(n)
#endif
#endif
#ifndef WAIT_ASYNCCNT
#define WAIT_ASYNCCNT(n) asm volatile("s_wait_asynccnt %0" ::"i"(n) : "memory")
#endif

#define AS_GLOBAL __attribute__((address_space(1)))
#define AS_LOCAL  __attribute__((address_space(3)))

// ---------------------------------------------------------------------------
// Weight convert + transpose: W[K,N] fp32 -> WT[N,K] bf16
// ---------------------------------------------------------------------------
__global__ void convert_transpose_kernel(const float* __restrict__ W,
                                         bf16_t* __restrict__ WT,
                                         int K, int N) {
    size_t t = (size_t)blockIdx.x * blockDim.x + threadIdx.x;
    if (t >= (size_t)K * N) return;
    int k = (int)(t / (size_t)N);
    int n = (int)(t % (size_t)N);
    WT[(size_t)n * K + k] = (bf16_t)W[t];
}

// ---------------------------------------------------------------------------
// Block reduction helper (wave32 shuffles + 8-wave LDS combine)
// ---------------------------------------------------------------------------
__device__ inline float block_reduce_sum_256(float p, float* red) {
#pragma unroll
    for (int off = 16; off > 0; off >>= 1) p += __shfl_xor(p, off, 32);
    int wid = threadIdx.x >> 5;
    if ((threadIdx.x & 31) == 0) red[wid] = p;
    __syncthreads();
    float s = 0.f;
#pragma unroll
    for (int w = 0; w < 8; ++w) s += red[w];
    __syncthreads();
    return s;
}

// ---------------------------------------------------------------------------
// Fused embedding gather + cross network. One block (256 thr) per batch row.
// ---------------------------------------------------------------------------
__global__ __launch_bounds__(256)
void embed_cross_kernel(const int*   __restrict__ ids,
                        const float* __restrict__ emb,
                        const float* __restrict__ cross_w,
                        const float* __restrict__ cross_b,
                        bf16_t* __restrict__ x0_bf,
                        float*  __restrict__ x_out) {
    __shared__ float x0s[DDIM];
    __shared__ float xs[DDIM];
    __shared__ float red[8];

    const int b = blockIdx.x;
    const int* idrow = ids + (size_t)b * NFIELD;

    for (int d = threadIdx.x; d < DDIM; d += 256) {
        int f = d >> 6;
        int e = d & 63;
        float v = emb[(size_t)idrow[f] * EDIM + e];
        x0s[d] = v;
        xs[d]  = v;
        x0_bf[(size_t)b * DDIM + d] = (bf16_t)v;
    }
    __syncthreads();

#pragma unroll
    for (int i = 0; i < 3; ++i) {
        float p = 0.f;
        for (int d = threadIdx.x; d < DDIM; d += 256)
            p += xs[d] * cross_w[(size_t)i * DDIM + d];
        float s = block_reduce_sum_256(p, red);
        for (int d = threadIdx.x; d < DDIM; d += 256)
            xs[d] = x0s[d] * s + cross_b[(size_t)i * DDIM + d] + xs[d];
        __syncthreads();
    }

    for (int d = threadIdx.x; d < DDIM; d += 256)
        x_out[(size_t)b * DDIM + d] = xs[d];
}

// ---------------------------------------------------------------------------
// bf16 WMMA GEMM:  C[M,N] = act(A[M,K] @ BT[N,K]^T + bias)
// Block tile 128x128x32, 256 threads = 8 waves, wave tile 32x64 (2x4 frags).
// Double-buffered LDS; global->LDS via async DMA (ASYNCcnt) when available.
// LDS row stride 40 halves (80 B): fragment b128 reads stay 16B-aligned.
// ---------------------------------------------------------------------------
template<bool RELU, bool OUT_BF16>
__global__ __launch_bounds__(256)
void gemm_bf16_wmma(const bf16_t* __restrict__ A,
                    const bf16_t* __restrict__ BT,
                    const float*  __restrict__ bias,
                    void* __restrict__ C,
                    int M, int N, int K) {
    constexpr int BM = 128, BN = 128, BK = 32;
    constexpr int AST = 40, BST = 40;            // LDS row stride in halves
    __shared__ bf16_t As[2][BM * AST];
    __shared__ bf16_t Bs[2][BN * BST];

    const int bm   = blockIdx.x * BM;
    const int bn   = blockIdx.y * BN;
    const int tid  = threadIdx.x;
    const int lane = tid & 31;
    const int wave = tid >> 5;
    const int wm   = (wave & 3) * 32;            // wave row offset in block
    const int wn   = (wave >> 2) * 64;           // wave col offset in block

    // Tile-load mapping: each thread moves 16 contiguous halves (two b128)
    // of A and of BT per K-step.
    const int lr  = tid >> 1;                    // 0..127 : tile row
    const int lsg = (tid & 1) * 16;              // 0 / 16 : half-row segment

    const bf16_t* gA = A  + (size_t)(bm + lr) * K + lsg;
    const bf16_t* gB = BT + (size_t)(bn + lr) * K + lsg;

    v8f acc[2][4];
#pragma unroll
    for (int mi = 0; mi < 2; ++mi)
#pragma unroll
        for (int ni = 0; ni < 4; ++ni)
#pragma unroll
            for (int e = 0; e < 8; ++e) acc[mi][ni][e] = 0.f;

    const int fm  = lane & 15;                   // frag row/col within tile
    const int kbA = (lane >> 4) * 16;            // A: K base {0,16}
    const int ksB = (lane >> 4) * 16;            // B: K base {0,16}

    // Stage a K-tile into LDS buffer `buf`.
    auto copy_tile = [&](int k0, int buf) {
        bf16_t* dA = &As[buf][lr * AST + lsg];
        bf16_t* dB = &Bs[buf][lr * BST + lsg];
#if USE_ASYNC_COPY
        __builtin_amdgcn_global_load_async_to_lds_b128(
            (AS_GLOBAL v4i*)(gA + k0),     (AS_LOCAL v4i*)(dA),     0, 0);
        __builtin_amdgcn_global_load_async_to_lds_b128(
            (AS_GLOBAL v4i*)(gA + k0 + 8), (AS_LOCAL v4i*)(dA + 8), 0, 0);
        __builtin_amdgcn_global_load_async_to_lds_b128(
            (AS_GLOBAL v4i*)(gB + k0),     (AS_LOCAL v4i*)(dB),     0, 0);
        __builtin_amdgcn_global_load_async_to_lds_b128(
            (AS_GLOBAL v4i*)(gB + k0 + 8), (AS_LOCAL v4i*)(dB + 8), 0, 0);
#else
        v8bf a0 = *(const v8bf*)(gA + k0);
        v8bf a1 = *(const v8bf*)(gA + k0 + 8);
        v8bf b0 = *(const v8bf*)(gB + k0);
        v8bf b1 = *(const v8bf*)(gB + k0 + 8);
        *(v8bf*)(dA)     = a0;
        *(v8bf*)(dA + 8) = a1;
        *(v8bf*)(dB)     = b0;
        *(v8bf*)(dB + 8) = b1;
#endif
    };

    const int nsteps = K / BK;
    copy_tile(0, 0);                             // prologue

    for (int i = 0; i < nsteps; ++i) {
        const int cur = i & 1;
        if (i + 1 < nsteps) {
            copy_tile((i + 1) * BK, cur ^ 1);    // overlap next tile DMA
#if USE_ASYNC_COPY
            WAIT_ASYNCCNT(4);                    // stage i's 4 copies landed
#endif
        } else {
#if USE_ASYNC_COPY
            WAIT_ASYNCCNT(0);
#endif
        }
#if !USE_ASYNC_COPY
        if (i + 2 < nsteps) {                    // warm L2/WGP for stage i+2
            __builtin_prefetch(gA + (i + 2) * BK, 0, 3);
            __builtin_prefetch(gB + (i + 2) * BK, 0, 3);
        }
#endif
        __syncthreads();                         // tile `cur` visible to all

        // Fragment loads, contiguous K order shared by A and B:
        //  A 16x32: lane half selects 16 contiguous K (0-15 vs 16-31)
        //  B 32x16: lane half selects 16 contiguous K (0-15 vs 16-31)
        v16bf afrag[2], bfrag[4];
#pragma unroll
        for (int mi = 0; mi < 2; ++mi) {
            const bf16_t* p = &As[cur][(wm + mi * 16 + fm) * AST + kbA];
            union { v16bf v; v8bf h[2]; } u;
            u.h[0] = *(const v8bf*)(p);
            u.h[1] = *(const v8bf*)(p + 8);
            afrag[mi] = u.v;
        }
#pragma unroll
        for (int ni = 0; ni < 4; ++ni) {
            const bf16_t* p = &Bs[cur][(wn + ni * 16 + fm) * BST + ksB];
            union { v16bf v; v8bf h[2]; } u;
            u.h[0] = *(const v8bf*)(p);
            u.h[1] = *(const v8bf*)(p + 8);
            bfrag[ni] = u.v;
        }
#pragma unroll
        for (int mi = 0; mi < 2; ++mi)
#pragma unroll
            for (int ni = 0; ni < 4; ++ni)
                acc[mi][ni] = __builtin_amdgcn_wmma_f32_16x16x32_bf16(
                    false, afrag[mi], false, bfrag[ni],
                    (short)0, acc[mi][ni], false, false);

        __syncthreads();                         // all reads of `cur` done
    }

    // Epilogue: bias + activation, write bf16 (for chaining) or fp32.
    const int rowoff = (lane >> 4) * 8;
#pragma unroll
    for (int mi = 0; mi < 2; ++mi) {
#pragma unroll
        for (int ni = 0; ni < 4; ++ni) {
            const int col   = bn + wn + ni * 16 + fm;
            const float bv  = bias[col];
            const int rbase = bm + wm + mi * 16 + rowoff;
#pragma unroll
            for (int r = 0; r < 8; ++r) {
                float v = acc[mi][ni][r] + bv;
                if (RELU) v = v > 0.f ? v : 0.f;
                if (OUT_BF16)
                    ((bf16_t*)C)[(size_t)(rbase + r) * N + col] = (bf16_t)v;
                else
                    ((float*)C)[(size_t)(rbase + r) * N + col] = v;
            }
        }
    }
}

// ---------------------------------------------------------------------------
// Output head: sigmoid(concat([x, h3]) @ out_w + out_b). Block per row.
// ---------------------------------------------------------------------------
__global__ __launch_bounds__(256)
void head_kernel(const float* __restrict__ x,
                 const float* __restrict__ h3,
                 const float* __restrict__ out_w,
                 const float* __restrict__ out_b,
                 float* __restrict__ out) {
    __shared__ float red[8];
    const int b = blockIdx.x;
    float p = 0.f;
    for (int d = threadIdx.x; d < DDIM; d += 256)
        p += x[(size_t)b * DDIM + d] * out_w[d];
    for (int d = threadIdx.x; d < H3D; d += 256)
        p += h3[(size_t)b * H3D + d] * out_w[DDIM + d];
    float s = block_reduce_sum_256(p, red);
    if (threadIdx.x == 0) {
        float z = s + out_b[0];
        out[b] = 1.f / (1.f + expf(-z));
    }
}

// ---------------------------------------------------------------------------
extern "C" void kernel_launch(void* const* d_in, const int* in_sizes, int n_in,
                              void* d_out, int out_size, void* d_ws, size_t ws_size,
                              hipStream_t stream) {
    const int*   ids     = (const int*)  d_in[0];
    const float* emb     = (const float*)d_in[1];
    const float* cross_w = (const float*)d_in[2];
    const float* cross_b = (const float*)d_in[3];
    const float* w1      = (const float*)d_in[4];
    const float* b1      = (const float*)d_in[5];
    const float* w2      = (const float*)d_in[6];
    const float* b2      = (const float*)d_in[7];
    const float* w3      = (const float*)d_in[8];
    const float* b3      = (const float*)d_in[9];
    const float* out_w   = (const float*)d_in[10];
    const float* out_b   = (const float*)d_in[11];
    float* out = (float*)d_out;

    // Carve workspace (256-byte aligned slabs).
    char* ws = (char*)d_ws;
    size_t off = 0;
    auto carve = [&](size_t bytes) -> void* {
        void* p = ws + off;
        off += (bytes + 255) & ~(size_t)255;
        return p;
    };
    bf16_t* x0_bf = (bf16_t*)carve((size_t)BATCH * DDIM * sizeof(bf16_t));
    float*  x_f   = (float*) carve((size_t)BATCH * DDIM * sizeof(float));
    bf16_t* w1T   = (bf16_t*)carve((size_t)H1D * DDIM * sizeof(bf16_t));
    bf16_t* w2T   = (bf16_t*)carve((size_t)H2D * H1D  * sizeof(bf16_t));
    bf16_t* w3T   = (bf16_t*)carve((size_t)H3D * H2D  * sizeof(bf16_t));
    bf16_t* h1    = (bf16_t*)carve((size_t)BATCH * H1D * sizeof(bf16_t));
    bf16_t* h2    = (bf16_t*)carve((size_t)BATCH * H2D * sizeof(bf16_t));
    float*  h3    = (float*) carve((size_t)BATCH * H3D * sizeof(float));
    (void)ws_size; (void)n_in; (void)in_sizes; (void)out_size;

    // 1) Weight convert + transpose (fp32 -> bf16, [K,N] -> [N,K]).
    {
        size_t n1 = (size_t)DDIM * H1D;
        size_t n2 = (size_t)H1D * H2D;
        size_t n3 = (size_t)H2D * H3D;
        convert_transpose_kernel<<<(unsigned)((n1 + 255) / 256), 256, 0, stream>>>(w1, w1T, DDIM, H1D);
        convert_transpose_kernel<<<(unsigned)((n2 + 255) / 256), 256, 0, stream>>>(w2, w2T, H1D, H2D);
        convert_transpose_kernel<<<(unsigned)((n3 + 255) / 256), 256, 0, stream>>>(w3, w3T, H2D, H3D);
    }

    // 2) Embedding gather + cross net.
    embed_cross_kernel<<<BATCH, 256, 0, stream>>>(ids, emb, cross_w, cross_b,
                                                  x0_bf, x_f);

    // 3-5) Deep tower GEMMs on the WMMA pipe.
    gemm_bf16_wmma<true,  true ><<<dim3(BATCH / 128, H1D / 128), 256, 0, stream>>>(
        x0_bf, w1T, b1, (void*)h1, BATCH, H1D, DDIM);
    gemm_bf16_wmma<true,  true ><<<dim3(BATCH / 128, H2D / 128), 256, 0, stream>>>(
        h1, w2T, b2, (void*)h2, BATCH, H2D, H1D);
    gemm_bf16_wmma<true,  false><<<dim3(BATCH / 128, H3D / 128), 256, 0, stream>>>(
        h2, w3T, b3, (void*)h3, BATCH, H3D, H2D);

    // 6) Output head.
    head_kernel<<<BATCH, 256, 0, stream>>>(x_f, h3, out_w, out_b, out);
}
